// CausalSelfAttention_627065225472
// MI455X (gfx1250) — matrix operand order repaired and checked by reading, not compiled, not run on hardware
//
#include <hip/hip_runtime.h>
#include <hip/hip_bf16.h>

// Problem constants (from reference)
#define BATCH 2
#define SEQ   2048
#define CH    1024
#define NH    16
#define HD    64
#define C3    (3 * CH)

typedef __attribute__((ext_vector_type(16))) __bf16 bf16x16;
typedef __attribute__((ext_vector_type(8)))  float  f32x8;
typedef __attribute__((ext_vector_type(4)))  unsigned int u32x4;
typedef __attribute__((ext_vector_type(8)))  unsigned int u32x8;

union FragAB {
    bf16x16 v;
    unsigned short u[16];
    uint4 q[2];
};
union FragC {
    f32x8 v;
    float f[8];
};

__device__ __forceinline__ unsigned short f2bf(float x) {
    unsigned int u = __float_as_uint(x);
    u += 0x7FFFu + ((u >> 16) & 1u);   // round-to-nearest-even
    return (unsigned short)(u >> 16);
}

__device__ __forceinline__ f32x8 wmma_bf16(const FragAB& a, const FragAB& b, f32x8 c) {
    return __builtin_amdgcn_wmma_f32_16x16x32_bf16(
        false, a.v, false, b.v, (short)0, c, false, false);
}

// --- DPP xor-butterfly reductions within a 16-lane row (no LDS traffic) ----
// xor1: quad_perm[1,0,3,2]=0xB1  xor2: quad_perm[2,3,0,1]=0x4E
// xor7: ROW_HALF_MIRROR=0x141    xor15: ROW_MIRROR=0x140
__device__ __forceinline__ float dpp_max16(float v) {
    int x = __float_as_int(v), y;
    y = __builtin_amdgcn_update_dpp(x, x, 0xB1, 0xF, 0xF, true);
    x = __float_as_int(fmaxf(__int_as_float(x), __int_as_float(y)));
    y = __builtin_amdgcn_update_dpp(x, x, 0x4E, 0xF, 0xF, true);
    x = __float_as_int(fmaxf(__int_as_float(x), __int_as_float(y)));
    y = __builtin_amdgcn_update_dpp(x, x, 0x141, 0xF, 0xF, true);
    x = __float_as_int(fmaxf(__int_as_float(x), __int_as_float(y)));
    y = __builtin_amdgcn_update_dpp(x, x, 0x140, 0xF, 0xF, true);
    return fmaxf(__int_as_float(x), __int_as_float(y));
}
__device__ __forceinline__ float dpp_sum16(float v) {
    int x = __float_as_int(v), y;
    y = __builtin_amdgcn_update_dpp(x, x, 0xB1, 0xF, 0xF, true);
    x = __float_as_int(__int_as_float(x) + __int_as_float(y));
    y = __builtin_amdgcn_update_dpp(x, x, 0x4E, 0xF, 0xF, true);
    x = __float_as_int(__int_as_float(x) + __int_as_float(y));
    y = __builtin_amdgcn_update_dpp(x, x, 0x141, 0xF, 0xF, true);
    x = __float_as_int(__int_as_float(x) + __int_as_float(y));
    y = __builtin_amdgcn_update_dpp(x, x, 0x140, 0xF, 0xF, true);
    return __int_as_float(x) + __int_as_float(y);
}

// ---------------------------------------------------------------------------
// fp32 -> bf16 bulk convert (vectorized x4)
// ---------------------------------------------------------------------------
__global__ void cvt_f32_bf16(const float* __restrict__ in,
                             unsigned short* __restrict__ out, int n4) {
    int i = blockIdx.x * blockDim.x + threadIdx.x;
    if (i >= n4) return;
    float4 f = ((const float4*)in)[i];
    ushort4 o;
    o.x = f2bf(f.x); o.y = f2bf(f.y); o.z = f2bf(f.z); o.w = f2bf(f.w);
    ((ushort4*)out)[i] = o;
}

// ---------------------------------------------------------------------------
// bf16 GEMM: out[M,N] = A[M,K] * B[K,N] + bias[N]
// Block tile 128x128, BK=32, 8 waves; wave tile 32x64 (2x4 WMMA frags).
// A tile moved with global_load_async_to_lds_b128 (ASYNCcnt path);
// B staged transposed in LDS so B-fragment reads are two contiguous 16B pieces.
// ---------------------------------------------------------------------------
template <bool OUT_BF16>
__global__ __launch_bounds__(256) void gemm_bf16(
    const unsigned short* __restrict__ A,   // [M,K] bf16
    const unsigned short* __restrict__ Bm,  // [K,N] bf16
    const float* __restrict__ bias,         // [N]
    void* __restrict__ out, int M, int N, int K) {

    __shared__ unsigned short As[128][40];  // [m][k], +8 pad
    __shared__ unsigned short Bs[128][40];  // [n][k] transposed, +8 pad

    const int tid  = threadIdx.x;
    const int lane = tid & 31;
    const int lo   = lane & 15;
    const int hi   = lane >> 4;
    const int w    = tid >> 5;
    const int wm   = w >> 1;
    const int wn   = w & 1;

    const int m0 = blockIdx.y * 128;
    const int n0 = blockIdx.x * 128;

    FragC acc[2][4];
#pragma unroll
    for (int i = 0; i < 2; i++)
#pragma unroll
        for (int j = 0; j < 4; j++)
#pragma unroll
            for (int r = 0; r < 8; r++) acc[i][j].f[r] = 0.0f;

    const int la_m = tid >> 1;        // 0..127
    const int la_k = (tid & 1) * 16;  // 0 or 16
    const int lb_k = tid >> 3;        // 0..31
    const int lb_n = (tid & 7) * 16;  // 0..112

    const unsigned ldsA = (unsigned)(uintptr_t)&As[la_m][la_k];

    for (int k0 = 0; k0 < K; k0 += 32) {
        // B tile into regs before the barrier (overlaps previous compute)
        const uint4* gb = (const uint4*)(Bm + (size_t)(k0 + lb_k) * N + n0 + lb_n);
        uint4 b0 = gb[0], b1 = gb[1];

        if (k0 + 32 < K) {  // global_prefetch_b8 hints for next tiles
            __builtin_prefetch(A + (size_t)(m0 + la_m) * K + k0 + 32 + la_k, 0, 0);
            __builtin_prefetch(Bm + (size_t)(k0 + 32 + lb_k) * N + n0 + lb_n, 0, 0);
        }

        __syncthreads();  // previous iteration's LDS reads done

        // A tile: async global -> LDS (32B per thread, two b128 ops)
        {
            unsigned long long gA = (unsigned long long)(uintptr_t)
                (A + (size_t)(m0 + la_m) * K + k0 + la_k);
            asm volatile("global_load_async_to_lds_b128 %0, %1, off"
                         :: "v"(ldsA), "v"(gA) : "memory");
            asm volatile("global_load_async_to_lds_b128 %0, %1, off"
                         :: "v"(ldsA + 16u), "v"(gA + 16ull) : "memory");
        }
        // B tile: transpose into Bs[n][k]
        {
            unsigned short bt[16];
            *(uint4*)&bt[0] = b0;
            *(uint4*)&bt[8] = b1;
#pragma unroll
            for (int i = 0; i < 16; i++) Bs[lb_n + i][lb_k] = bt[i];
        }
        asm volatile("s_wait_asynccnt 0" ::: "memory");
        __syncthreads();  // tiles visible

        FragAB af[2];
#pragma unroll
        for (int mi = 0; mi < 2; mi++) {
            const int row = wm * 32 + mi * 16 + lo;
            af[mi].q[0] = *(const uint4*)&As[row][hi * 8];
            af[mi].q[1] = *(const uint4*)&As[row][16 + hi * 8];
        }
#pragma unroll
        for (int ni = 0; ni < 4; ni++) {
            FragAB bf;
            const int col = wn * 64 + ni * 16 + lo;
            bf.q[0] = *(const uint4*)&Bs[col][hi * 8];
            bf.q[1] = *(const uint4*)&Bs[col][16 + hi * 8];
#pragma unroll
            for (int mi = 0; mi < 2; mi++)
                acc[mi][ni].v = wmma_bf16(af[mi], bf, acc[mi][ni].v);
        }
    }

#pragma unroll
    for (int ni = 0; ni < 4; ni++) {
        const int col = n0 + wn * 64 + ni * 16 + lo;
        const float bv = bias[col];
#pragma unroll
        for (int mi = 0; mi < 2; mi++) {
#pragma unroll
            for (int r = 0; r < 8; r++) {
                const int row = m0 + wm * 32 + mi * 16 + r + 8 * hi;
                const float v = acc[mi][ni].f[r] + bv;
                if (OUT_BF16)
                    ((unsigned short*)out)[(size_t)row * N + col] = f2bf(v);
                else
                    ((float*)out)[(size_t)row * N + col] = v;
            }
        }
    }
}

// ---------------------------------------------------------------------------
// Flash attention forward. qkv: [B,T,3C] bf16.
// K tile fetched by the Tensor Data Mover (tensor_load_to_lds, TENSORcnt),
// with D# LDS padding reproducing the [32][72] bank-conflict-free layout.
// Softmax row reductions use DPP xor-butterflies (VALU, co-issues with WMMA).
// ---------------------------------------------------------------------------
__global__ __launch_bounds__(256) void attn_fwd(
    const unsigned short* __restrict__ qkv,  // [B,T,3C] bf16
    unsigned short* __restrict__ y) {        // [B,T,C]  bf16

    __shared__ unsigned short Ks[32][72];     // [key][d], +8 pad (TDM pad fills this)
    __shared__ unsigned short Vt[64][40];     // [d][key] transposed, +8 pad
    __shared__ unsigned short Ps[8][16][40];  // per-wave P staging

    const int tid  = threadIdx.x;
    const int lane = tid & 31;
    const int lo   = lane & 15;
    const int hi   = lane >> 4;
    const int w    = tid >> 5;

    const int qtile = blockIdx.x;
    const int bh    = blockIdx.y;
    const int b     = bh / NH;
    const int h     = bh % NH;

    const int qbase = qtile * 128;
    const int qrow0 = qbase + w * 16;

    // Q fragments, resident for the whole kernel (A layout: lane = row)
    FragAB qf[2];
    {
        const size_t base = (size_t)(b * SEQ + qrow0 + lo) * C3 + h * HD;
        const uint4* gq = (const uint4*)(qkv + base);
#pragma unroll
        for (int f = 0; f < 2; f++) {
            qf[f].q[0] = gq[(f * 32 + hi * 8) / 8];
            qf[f].q[1] = gq[(f * 32 + 16 + hi * 8) / 8];
        }
    }

    FragC o[4];
#pragma unroll
    for (int f = 0; f < 4; f++)
#pragma unroll
        for (int r = 0; r < 8; r++) o[f].f[r] = 0.0f;
    float mrow[8], lrow[8];
#pragma unroll
    for (int r = 0; r < 8; r++) { mrow[r] = -3.0e38f; lrow[r] = 0.0f; }

    const float sc = 0.125f;  // 1/sqrt(64)

    // ---- TDM descriptor (constant part), per ISA cdna5 8.3/8.4 ----
    // 2D tile: 32 rows (keys) x 64 cols (d), data_size=2B.
    // pad_enable, pad_interval=4 (32 DWORDs = one 128B row), pad_amount=3
    // (4 DWORDs = 16B) -> hardware writes the [32][72] padded layout.
    u32x8 tg1;
    tg1[0] = (1u << 16) | (1u << 20) | (4u << 22) | (3u << 25); // data_size=2B, pad
    tg1[1] = (unsigned)(C3 & 0xFFFF) << 16;                     // tensor_dim0 lo
    tg1[2] = ((unsigned)(BATCH * SEQ) & 0xFFFF) << 16;          // dim0 hi=0 | tensor_dim1 lo
    tg1[3] = (unsigned)HD << 16;                                // tensor_dim1 hi=0 | tile_dim0=64
    tg1[4] = 32u;                                               // tile_dim1=32, tile_dim2=0
    tg1[5] = (unsigned)C3;                                      // tensor_dim0_stride = 3072
    tg1[6] = 0u;
    tg1[7] = 0u;
    u32x4 tgz;
    tgz[0] = 0u; tgz[1] = 0u; tgz[2] = 0u; tgz[3] = 0u;         // groups 2/3 (dims 2..4 unused)
    const unsigned ksLds = (unsigned)(uintptr_t)&Ks[0][0];

    const int nsteps = (qbase + 128) / 32;
    const int ldk = tid >> 3;               // 0..31 : key row
    const int ldd = (tid & 7) * 8;          // 0..56 : d chunk

    for (int s = 0; s < nsteps; s++) {
        const int kbase = s * 32;
        // V tile into regs before the barrier
        const size_t vb = (size_t)(b * SEQ + kbase + ldk) * C3 + 2 * CH + h * HD + ldd;
        uint4 v4 = *(const uint4*)(qkv + vb);

        __syncthreads();  // previous step's LDS reads done

        if (w == 0) {  // one wave drives the TDM (EXEC ignored by tensor ops)
            unsigned long long ga = (unsigned long long)(uintptr_t)qkv +
                2ull * ((size_t)(b * SEQ + kbase) * C3 + CH + h * HD);
            u32x4 tg0;
            tg0.x = 1u;                                    // count=1, user D#
            tg0.y = ksLds;                                 // lds_addr
            tg0.z = (unsigned)ga;                          // global_addr lo
            tg0.w = (unsigned)(ga >> 32) | (2u << 30);     // global_addr hi | type=2
            asm volatile("tensor_load_to_lds %0, %1, %2, %3"
                         :: "s"(tg0), "s"(tg1), "s"(tgz), "s"(tgz) : "memory");
        }
        // V tile: transpose into Vt[d][key]
        {
            unsigned short vt[8];
            *(uint4*)&vt[0] = v4;
#pragma unroll
            for (int i = 0; i < 8; i++) Vt[ldd + i][ldk] = vt[i];
        }
        if (w == 0) __builtin_amdgcn_s_wait_tensorcnt(0);
        __syncthreads();  // K (TDM) + V visible to all waves

        // S = Q K^T for key groups n=0,1 (16 keys each)
        FragC sacc[2];
#pragma unroll
        for (int n = 0; n < 2; n++) {
#pragma unroll
            for (int r = 0; r < 8; r++) sacc[n].f[r] = 0.0f;
            const int key = n * 16 + lo;  // B layout: lane = key column
#pragma unroll
            for (int f = 0; f < 2; f++) {
                FragAB kf;
                kf.q[0] = *(const uint4*)&Ks[key][f * 32 + hi * 8];
                kf.q[1] = *(const uint4*)&Ks[key][f * 32 + 16 + hi * 8];
                sacc[n].v = wmma_bf16(qf[f], kf, sacc[n].v);
            }
        }

        // Online softmax; row (r + 8*hi) lives in one 16-lane DPP row
#pragma unroll
        for (int r = 0; r < 8; r++) {
            const int q   = qrow0 + r + 8 * hi;
            const bool v0 = (kbase + lo <= q);
            const bool v1 = (kbase + 16 + lo <= q);
            float s0 = v0 ? sacc[0].f[r] * sc : -3.0e38f;
            float s1 = v1 ? sacc[1].f[r] * sc : -3.0e38f;
            const float mx   = dpp_max16(fmaxf(s0, s1));
            const float mnew = fmaxf(mrow[r], mx);
            const float scl  = __expf(mrow[r] - mnew);
            float p0 = v0 ? __expf(s0 - mnew) : 0.0f;
            float p1 = v1 ? __expf(s1 - mnew) : 0.0f;
            const float rs = dpp_sum16(p0 + p1);
            lrow[r] = lrow[r] * scl + rs;
            mrow[r] = mnew;
#pragma unroll
            for (int f = 0; f < 4; f++) o[f].f[r] *= scl;
            Ps[w][r + 8 * hi][lo]      = f2bf(p0);
            Ps[w][r + 8 * hi][16 + lo] = f2bf(p1);
        }

        // wave-local LDS fence: P writes -> P fragment reads
        asm volatile("s_wait_dscnt 0" ::: "memory");

        FragAB pf;  // A layout: lane = q row, K = 32 keys
        pf.q[0] = *(const uint4*)&Ps[w][lo][hi * 8];
        pf.q[1] = *(const uint4*)&Ps[w][lo][16 + hi * 8];

#pragma unroll
        for (int f = 0; f < 4; f++) {  // V: B layout, lane = d column
            FragAB vf;
            const int dcol = f * 16 + lo;
            vf.q[0] = *(const uint4*)&Vt[dcol][hi * 8];
            vf.q[1] = *(const uint4*)&Vt[dcol][16 + hi * 8];
            o[f].v = wmma_bf16(pf, vf, o[f].v);
        }
    }

    // normalize + write y[b,t,h,d] (== [B,T,C])
#pragma unroll
    for (int r = 0; r < 8; r++) {
        const float inv = 1.0f / lrow[r];
        const int t = qrow0 + r + 8 * hi;
        const size_t ob = (size_t)(b * SEQ + t) * CH + h * HD;
#pragma unroll
        for (int f = 0; f < 4; f++)
            y[ob + f * 16 + lo] = f2bf(o[f].f[r] * inv);
    }
}

// ---------------------------------------------------------------------------
extern "C" void kernel_launch(void* const* d_in, const int* in_sizes, int n_in,
                              void* d_out, int out_size, void* d_ws, size_t ws_size,
                              hipStream_t stream) {
    const float* x     = (const float*)d_in[0];  // [B,T,C]
    const float* Wqkv  = (const float*)d_in[1];  // [C,3C]
    const float* bqkv  = (const float*)d_in[2];  // [3C]
    const float* Wproj = (const float*)d_in[3];  // [C,C]
    const float* bproj = (const float*)d_in[4];  // [C]
    float* out = (float*)d_out;                  // [B,T,C] fp32

    const int M = BATCH * SEQ;  // 4096

    // workspace layout (bf16 buffers), total ~50.3 MB
    char* ws = (char*)d_ws;
    unsigned short* xb    = (unsigned short*)ws;  ws += (size_t)M * CH * 2;
    unsigned short* wqkvb = (unsigned short*)ws;  ws += (size_t)CH * C3 * 2;
    unsigned short* wprjb = (unsigned short*)ws;  ws += (size_t)CH * CH * 2;
    unsigned short* qkvb  = (unsigned short*)ws;  ws += (size_t)M * C3 * 2;
    unsigned short* yb    = (unsigned short*)ws;  ws += (size_t)M * CH * 2;

    {
        int n4 = M * CH / 4;
        cvt_f32_bf16<<<(n4 + 255) / 256, 256, 0, stream>>>(x, xb, n4);
        n4 = CH * C3 / 4;
        cvt_f32_bf16<<<(n4 + 255) / 256, 256, 0, stream>>>(Wqkv, wqkvb, n4);
        n4 = CH * CH / 4;
        cvt_f32_bf16<<<(n4 + 255) / 256, 256, 0, stream>>>(Wproj, wprjb, n4);
    }

    gemm_bf16<true><<<dim3(C3 / 128, M / 128), 256, 0, stream>>>(
        xb, wqkvb, bqkv, qkvb, M, C3, CH);

    attn_fwd<<<dim3(SEQ / 128, BATCH * NH), 256, 0, stream>>>(qkvb, yb);

    gemm_bf16<false><<<dim3(CH / 128, M / 128), 256, 0, stream>>>(
        yb, wprjb, bproj, out, M, CH, CH);
}
